// ScGPTEncoder_54322746359910
// MI455X (gfx1250) — matrix operand order repaired and checked by reading, not compiled, not run on hardware
//
#include <hip/hip_runtime.h>
#include <hip/hip_bf16.h>
#include <math.h>

#define Bb   32
#define Cc   2
#define Ll   17819
#define Hh   256
#define NLy  6
#define NHh  8
#define MLPd 512
#define Tt   512
#define SP   528          // padded seq len (33 * 16)
#define BINS 51

typedef __attribute__((ext_vector_type(16))) __bf16 bf16x16;
typedef __attribute__((ext_vector_type(8)))  __bf16 bf16x8;
typedef __attribute__((ext_vector_type(8)))  float  f32x8;
typedef __attribute__((ext_vector_type(4)))  unsigned int u32x4;
typedef __attribute__((ext_vector_type(8)))  int    i32x8;
typedef __attribute__((ext_vector_type(4)))  int    i32x4;

#if defined(__has_builtin)
#if __has_builtin(__builtin_amdgcn_tensor_load_to_lds) && \
    __has_builtin(__builtin_amdgcn_s_wait_tensorcnt)
#define USE_TDM 1
#endif
#endif

static __device__ __forceinline__ f32x8 wmma_bf16(bf16x16 a, bf16x16 b, f32x8 c) {
  // D = A(16x32 bf16) * B(32x16 bf16) + C(16x16 f32)
  return __builtin_amdgcn_wmma_f32_16x16x32_bf16(false, a, false, b, (short)0, c, false, false);
}

// A-fragment: row-major bf16 row; lane holds row, K = 16*half + j
static __device__ __forceinline__ bf16x16 load_a_frag(const __bf16* row, int k0, int half) {
  bf16x8 lo = *(const bf16x8*)(row + k0 + 16 * half);
  bf16x8 hi = *(const bf16x8*)(row + k0 + 16 * half + 8);
  return __builtin_shufflevector(lo, hi, 0,1,2,3,4,5,6,7,8,9,10,11,12,13,14,15);
}
// B-fragment: row-major bf16 row (W[n][k]); lane holds col n, K = 16*half + j
static __device__ __forceinline__ bf16x16 load_b_frag(const __bf16* row, int k0, int half) {
  const bf16x8* p = (const bf16x8*)(row + k0 + 16 * half);
  bf16x8 lo = p[0], hi = p[1];
  return __builtin_shufflevector(lo, hi, 0,1,2,3,4,5,6,7,8,9,10,11,12,13,14,15);
}
// A-matrix per-lane K index (contiguous K = 16*half + j), used for scalar gathers
static __device__ __forceinline__ int a_kmap(int j, int half) {
  int g = j >> 1, p = j & 1;
  return 16 * half + j;
}

// ---------------------------------------------------------------------------
// Phase 1: exact per-row quantile binning (stable rank among positives)
// ---------------------------------------------------------------------------
__global__ void bin_kernel(const float* __restrict__ x, int* __restrict__ bins) {
  int row = blockIdx.x;                       // 0 .. B*C-1
  const float* xr = x + (size_t)row * Ll;
  __shared__ int s_cnt;
  if (threadIdx.x == 0) s_cnt = 0;
  __syncthreads();
  int local = 0;
  for (int j = threadIdx.x; j < Ll; j += blockDim.x)
    if (xr[j] > 0.f) local++;
  atomicAdd(&s_cnt, local);
  __syncthreads();
  float cnt = (float)(s_cnt > 0 ? s_cnt : 1);
  int* br = bins + (size_t)row * Ll;
  for (int i = threadIdx.x; i < Ll; i += blockDim.x) {
    float vi = xr[i];
    if (!(vi > 0.f)) { br[i] = 0; continue; }
    int rank = 0;
    for (int j = 0; j < Ll; ++j) {
      float vj = xr[j];
      if (vj > 0.f && (vj < vi || (vj == vi && j < i))) rank++;
    }
    int bin = (int)ceilf((float)(rank + 1) * (float)BINS / cnt);
    br[i] = bin < 1 ? 1 : (bin > BINS ? BINS : bin);
  }
}

// ---------------------------------------------------------------------------
// Phase 2: exact top-512 (value desc, index asc) via counting sort, 1 blk / b
// ---------------------------------------------------------------------------
__global__ void topk_kernel(const int* __restrict__ bins,
                            int* __restrict__ idx, int* __restrict__ gene,
                            int* __restrict__ selb, float* __restrict__ maskb) {
  int b = blockIdx.x;
  const int* b0 = bins + (size_t)(b * 2 + 0) * Ll;
  const int* b1 = bins + (size_t)(b * 2 + 1) * Ll;
  __shared__ int hist[103];
  __shared__ int seen[103];
  __shared__ int cg[103];
  for (int v = threadIdx.x; v < 103; v += blockDim.x) { hist[v] = 0; seen[v] = 0; }
  __syncthreads();
  for (int l = threadIdx.x; l < Ll; l += blockDim.x)
    atomicAdd(&hist[b0[l] + b1[l]], 1);
  for (int t = threadIdx.x; t < Tt; t += blockDim.x) {
    idx[b * Tt + t] = 0; gene[b * Tt + t] = 0;
    selb[(b * 2 + 0) * Tt + t] = 0; selb[(b * 2 + 1) * Tt + t] = 0;
  }
  for (int s = threadIdx.x; s < SP; s += blockDim.x)
    maskb[b * SP + s] = (s == 0) ? 0.f : -1e9f;
  __syncthreads();
  if (threadIdx.x == 0) {
    int run = 0;
    for (int v = 102; v >= 0; --v) { cg[v] = run; run += hist[v]; }
    int nsel = 0;
    for (int l = 0; l < Ll && nsel < Tt; ++l) {
      int v = b0[l] + b1[l];
      if (v < 1) continue;
      int slot = cg[v] + seen[v]; seen[v]++;
      if (slot < Tt) {
        idx[b * Tt + slot] = l;
        gene[b * Tt + slot] = l + 1;
        selb[(b * 2 + 0) * Tt + slot] = b0[l];
        selb[(b * 2 + 1) * Tt + slot] = b1[l];
        maskb[b * SP + 1 + slot] = 0.f;
        nsel++;
      }
    }
    if (nsel == 0) {  // empty-sample fallback
      idx[b * Tt] = 0; gene[b * Tt] = 1;
      selb[(b * 2 + 0) * Tt] = b0[0]; selb[(b * 2 + 1) * Tt] = b1[0];
      maskb[b * SP + 1] = 0.f;
    }
  }
}

// ---------------------------------------------------------------------------
// Phase 3a: token build  h[b,s,:] f32 ; pad rows (s>=513) zeroed
// ---------------------------------------------------------------------------
__global__ void token_kernel(const int* __restrict__ gene, const int* __restrict__ selb,
                             const float* __restrict__ gene_embed,
                             const float* __restrict__ vw, const float* __restrict__ vb,
                             const float* __restrict__ cs, const float* __restrict__ cls,
                             float* __restrict__ hbuf) {
  int s = blockIdx.x, b = blockIdx.y, hh = threadIdx.x;
  float v;
  if (s == 0) {
    v = cls[hh];
  } else if (s <= Tt) {
    int t = s - 1;
    int g = gene[b * Tt + t];
    v = gene_embed[(size_t)g * Hh + hh];
    #pragma unroll
    for (int c = 0; c < Cc; ++c) {
      float sc = cs[c * Hh + hh];
      v += ((float)selb[(b * 2 + c) * Tt + t] * (1.0f / (float)BINS)) * (vw[c * Hh + hh] * sc);
      v += vb[c * Hh + hh] * sc;
    }
  } else {
    v = 0.f;
  }
  hbuf[((size_t)b * SP + s) * Hh + hh] = v;
}

// ---------------------------------------------------------------------------
// f32 -> bf16 weight conversion (grid-stride)
// ---------------------------------------------------------------------------
__global__ void cvt_kernel(const float* __restrict__ in, __bf16* __restrict__ out, int n) {
  for (int i = blockIdx.x * blockDim.x + threadIdx.x; i < n; i += gridDim.x * blockDim.x)
    out[i] = (__bf16)in[i];
}

// ---------------------------------------------------------------------------
// LayerNorm over H=256, f32 in -> bf16 out, one block per row
// ---------------------------------------------------------------------------
__global__ void ln_kernel(const float* __restrict__ in, const float* __restrict__ sg,
                          const float* __restrict__ bg, __bf16* __restrict__ out) {
  __shared__ float red[256];
  int row = blockIdx.x, t = threadIdx.x;
  float x = in[(size_t)row * Hh + t];
  red[t] = x; __syncthreads();
  for (int o = 128; o > 0; o >>= 1) { if (t < o) red[t] += red[t + o]; __syncthreads(); }
  float mean = red[0] * (1.0f / Hh);
  __syncthreads();
  float d = x - mean;
  red[t] = d * d; __syncthreads();
  for (int o = 128; o > 0; o >>= 1) { if (t < o) red[t] += red[t + o]; __syncthreads(); }
  float var = red[0] * (1.0f / Hh);
  out[(size_t)row * Hh + t] = (__bf16)(d * rsqrtf(var + 1e-5f) * sg[t] + bg[t]);
}

__global__ void final_ln_kernel(const float* __restrict__ h, const float* __restrict__ sg,
                                const float* __restrict__ bg, float* __restrict__ out) {
  __shared__ float red[256];
  int b = blockIdx.x, t = threadIdx.x;
  float x = h[((size_t)b * SP) * Hh + t];       // CLS row only
  red[t] = x; __syncthreads();
  for (int o = 128; o > 0; o >>= 1) { if (t < o) red[t] += red[t + o]; __syncthreads(); }
  float mean = red[0] * (1.0f / Hh);
  __syncthreads();
  float d = x - mean;
  red[t] = d * d; __syncthreads();
  for (int o = 128; o > 0; o >>= 1) { if (t < o) red[t] += red[t + o]; __syncthreads(); }
  float var = red[0] * (1.0f / Hh);
  out[b * Hh + t] = d * rsqrtf(var + 1e-5f) * sg[t] + bg[t];
}

// ---------------------------------------------------------------------------
// WMMA GEMM: out[b,m,n] = sum_k A[b,m,k]*W[n,k] (+bias)(gelu?)(+res)
// A,W bf16 row-major (128-bit fragment loads); one wave = 16(M) x 64(N) tile
// ---------------------------------------------------------------------------
__global__ void gemm_kernel(const __bf16* __restrict__ A, int strideA, int lda,
                            const __bf16* __restrict__ W, int ldw,
                            const float* __restrict__ bias,
                            const float* __restrict__ res, int strideO,
                            float* __restrict__ outF, __bf16* __restrict__ outB,
                            int ldo, int K, int act) {
  int lane = threadIdx.x;
  int half = lane >> 4, mr = lane & 15, nc = lane & 15;
  int m0 = blockIdx.x * 16, n0 = blockIdx.y * 64, b = blockIdx.z;
  const __bf16* Ar = A + (size_t)b * strideA + (size_t)(m0 + mr) * lda;
  const __bf16* Wr = W + (size_t)(n0 + nc) * ldw;
  f32x8 acc[4] = {{}, {}, {}, {}};
  for (int k0 = 0; k0 < K; k0 += 32) {
    bf16x16 af = load_a_frag(Ar, k0, half);
    #pragma unroll
    for (int nt = 0; nt < 4; ++nt) {
      bf16x16 wf = load_b_frag(Wr + (size_t)(nt * 16) * ldw, k0, half);
      acc[nt] = wmma_bf16(af, wf, acc[nt]);
    }
  }
  #pragma unroll
  for (int nt = 0; nt < 4; ++nt) {
    #pragma unroll
    for (int r = 0; r < 8; ++r) {
      int m = m0 + r + 8 * half, n = n0 + nt * 16 + nc;
      float v = acc[nt][r];
      if (bias) v += bias[n];
      if (act == 1) v = 0.5f * v * (1.0f + erff(v * 0.70710678118654752f));
      size_t off = (size_t)b * strideO + (size_t)m * ldo + n;
      if (res) v += res[off];
      if (outF) outF[off] = v;
      else      outB[off] = (__bf16)v;
    }
  }
}

// ---------------------------------------------------------------------------
// Flash attention: one wave per (query-tile, head, batch). dh=32 -> 1 WMMA QK,
// 2 WMMA PV per key tile; V tile staged into LDS via Tensor Data Mover
// ---------------------------------------------------------------------------
__global__ void attn_kernel(const __bf16* __restrict__ qkv, const float* __restrict__ maskb,
                            __bf16* __restrict__ att) {
  int lane = threadIdx.x;
  int half = lane >> 4, mr = lane & 15, nc = lane & 15;
  int qt = blockIdx.x, head = blockIdx.y, b = blockIdx.z;
  const __bf16* qkvb = qkv + (size_t)b * SP * 768;
  const float* mb = maskb + b * SP;
  const float scale = 0.17677669529663687f;   // 1/sqrt(32)

  __shared__ __bf16 vtile[16][32];            // V key tile (16 keys x dh=32)
  __shared__ float pt[16][17];                // P transpose staging

  bf16x16 qa = load_a_frag(qkvb + (size_t)(qt * 16 + mr) * 768 + head * 32, 0, half);

  float mrun[8], lrun[8];
  f32x8 acc0 = {}, acc1 = {};
  #pragma unroll
  for (int r = 0; r < 8; ++r) { mrun[r] = -1e30f; lrun[r] = 0.f; }

  for (int kt = 0; kt < SP / 16; ++kt) {
    // --- Q @ K^T -----------------------------------------------------------
    bf16x16 kf = load_b_frag(qkvb + (size_t)(kt * 16 + nc) * 768 + 256 + head * 32, 0, half);
    f32x8 sc = {};
    sc = wmma_bf16(qa, kf, sc);

    // --- online softmax ----------------------------------------------------
    float kmask = mb[kt * 16 + nc];
    float p[8];
    #pragma unroll
    for (int r = 0; r < 8; ++r) {
      float s = sc[r] * scale + kmask;
      float rmax = s;
      for (int o = 8; o >= 1; o >>= 1) rmax = fmaxf(rmax, __shfl_xor(rmax, o, 32));
      float nm = fmaxf(mrun[r], rmax);
      float pe = __expf(s - nm);
      float rsum = pe;
      for (int o = 8; o >= 1; o >>= 1) rsum += __shfl_xor(rsum, o, 32);
      float alpha = __expf(mrun[r] - nm);
      lrun[r] = lrun[r] * alpha + rsum;
      mrun[r] = nm;
      acc0[r] *= alpha; acc1[r] *= alpha;
      p[r] = pe;
    }
    // --- transpose P (C layout -> A layout) through LDS --------------------
    __syncthreads();
    #pragma unroll
    for (int r = 0; r < 8; ++r) pt[r + 8 * half][nc] = p[r];
    __syncthreads();
    bf16x16 pa;
    #pragma unroll
    for (int j = 0; j < 16; ++j) {
      int kk = a_kmap(j, half);
      pa[j] = (kk < 16) ? (__bf16)pt[mr][kk] : (__bf16)0.0f;
    }
    // --- stage V tile (16 x 32 bf16, row stride 768) into LDS --------------
    const __bf16* vsrc = qkvb + (size_t)(kt * 16) * 768 + 512 + head * 32;
    __syncthreads();
#ifdef USE_TDM
    {
      unsigned long long ga = (unsigned long long)(uintptr_t)vsrc;
      unsigned ldsa = (unsigned)(uintptr_t)(&vtile[0][0]);
      // D# group0: count=1 | lds_addr | global_addr | type=2 (bits 127:126)
      u32x4 g0 = { 1u, ldsa, (unsigned)ga,
                   (unsigned)((ga >> 32) & 0x1FFFFFFull) | (2u << 30) };
      // D# group1: data_size=2B; tensor_dim0=32; tensor_dim1=16; tile_dim0=32;
      //            tile_dim1=16; tensor_dim0_stride=768
      i32x8 g1 = { (int)(1u << 16), (int)(32u << 16), (int)(16u << 16),
                   (int)(32u << 16), 16, 768, 0, 0 };
      i32x4 gz4 = { 0, 0, 0, 0 };
      i32x8 gz8 = { 0, 0, 0, 0, 0, 0, 0, 0 };
      __builtin_amdgcn_tensor_load_to_lds(g0, g1, gz4, gz4, gz8, 0);
      __builtin_amdgcn_s_wait_tensorcnt(0);
    }
#else
    #pragma unroll
    for (int i = 0; i < 2; ++i) {
      int c = lane + 32 * i;                  // 64 chunks of 8 bf16
      int row = c >> 2, c8 = c & 3;
      *(bf16x8*)(&vtile[row][c8 * 8]) = *(const bf16x8*)(vsrc + (size_t)row * 768 + c8 * 8);
    }
#endif
    __syncthreads();
    // --- P @ V (B layout from LDS; K rows 16..31 zero, matching P cols) ----
    bf16x16 v0f, v1f;
    #pragma unroll
    for (int j = 0; j < 16; ++j) {
      if (half == 0) { v0f[j] = vtile[j][nc]; v1f[j] = vtile[j][16 + nc]; }
      else           { v0f[j] = (__bf16)0.0f; v1f[j] = (__bf16)0.0f; }
    }
    acc0 = wmma_bf16(pa, v0f, acc0);
    acc1 = wmma_bf16(pa, v1f, acc1);
  }

  __bf16* ob = att + (size_t)b * SP * Hh;
  #pragma unroll
  for (int r = 0; r < 8; ++r) {
    int m = qt * 16 + r + 8 * half;
    float inv = 1.0f / lrun[r];
    ob[(size_t)m * Hh + head * 32 + nc]      = (__bf16)(acc0[r] * inv);
    ob[(size_t)m * Hh + head * 32 + 16 + nc] = (__bf16)(acc1[r] * inv);
  }
}

// ---------------------------------------------------------------------------
extern "C" void kernel_launch(void* const* d_in, const int* in_sizes, int n_in,
                              void* d_out, int out_size, void* d_ws, size_t ws_size,
                              hipStream_t stream) {
  const float* x          = (const float*)d_in[0];
  const float* gene_embed = (const float*)d_in[1];
  const float* value_w    = (const float*)d_in[2];
  const float* value_b    = (const float*)d_in[3];
  const float* cscale     = (const float*)d_in[4];
  const float* cls        = (const float*)d_in[5];
  const float* in_proj_w  = (const float*)d_in[6];
  const float* in_proj_b  = (const float*)d_in[7];
  const float* out_proj_w = (const float*)d_in[8];
  const float* out_proj_b = (const float*)d_in[9];
  const float* ln1_s = (const float*)d_in[10];
  const float* ln1_b = (const float*)d_in[11];
  const float* ln2_s = (const float*)d_in[12];
  const float* ln2_b = (const float*)d_in[13];
  const float* ffn_w1 = (const float*)d_in[14];
  const float* ffn_b1 = (const float*)d_in[15];
  const float* ffn_w2 = (const float*)d_in[16];
  const float* ffn_b2 = (const float*)d_in[17];
  const float* norm_s = (const float*)d_in[18];
  const float* norm_b = (const float*)d_in[19];

  char* ws = (char*)d_ws;
  size_t off = 0;
  int*    bins   = (int*)(ws + off);    off += (size_t)Bb * Cc * Ll * 4;
  int*    idx    = (int*)(ws + off);    off += (size_t)Bb * Tt * 4;
  int*    gene   = (int*)(ws + off);    off += (size_t)Bb * Tt * 4;
  int*    selb   = (int*)(ws + off);    off += (size_t)Bb * Cc * Tt * 4;
  float*  maskb  = (float*)(ws + off);  off += (size_t)Bb * SP * 4;
  float*  h      = (float*)(ws + off);  off += (size_t)Bb * SP * Hh * 4;
  __bf16* hn     = (__bf16*)(ws + off); off += (size_t)Bb * SP * Hh * 2;
  __bf16* qkvb   = (__bf16*)(ws + off); off += (size_t)Bb * SP * 768 * 2;
  __bf16* attb   = (__bf16*)(ws + off); off += (size_t)Bb * SP * Hh * 2;
  __bf16* ffb    = (__bf16*)(ws + off); off += (size_t)Bb * SP * MLPd * 2;
  __bf16* wqkv   = (__bf16*)(ws + off); off += (size_t)NLy * 768 * Hh * 2;
  __bf16* wout   = (__bf16*)(ws + off); off += (size_t)NLy * Hh * Hh * 2;
  __bf16* wf1    = (__bf16*)(ws + off); off += (size_t)NLy * MLPd * Hh * 2;
  __bf16* wf2    = (__bf16*)(ws + off); off += (size_t)NLy * Hh * MLPd * 2;

  // one-time weight conversion to bf16
  {
    int n1 = NLy * 768 * Hh, n2 = NLy * Hh * Hh, n3 = NLy * MLPd * Hh, n4 = NLy * Hh * MLPd;
    cvt_kernel<<<(n1 + 255) / 256, 256, 0, stream>>>(in_proj_w,  wqkv, n1);
    cvt_kernel<<<(n2 + 255) / 256, 256, 0, stream>>>(out_proj_w, wout, n2);
    cvt_kernel<<<(n3 + 255) / 256, 256, 0, stream>>>(ffn_w1,     wf1,  n3);
    cvt_kernel<<<(n4 + 255) / 256, 256, 0, stream>>>(ffn_w2,     wf2,  n4);
  }

  bin_kernel<<<Bb * Cc, 256, 0, stream>>>(x, bins);
  topk_kernel<<<Bb, 256, 0, stream>>>(bins, idx, gene, selb, maskb);
  token_kernel<<<dim3(SP, Bb), 256, 0, stream>>>(gene, selb, gene_embed, value_w,
                                                 value_b, cscale, cls, h);

  const int MT = SP / 16;   // 33 row tiles
  for (int l = 0; l < NLy; ++l) {
    ln_kernel<<<Bb * SP, 256, 0, stream>>>(h, ln1_s + l * Hh, ln1_b + l * Hh, hn);
    gemm_kernel<<<dim3(MT, 768 / 64, Bb), 32, 0, stream>>>(
        hn, SP * Hh, Hh, wqkv + (size_t)l * 768 * Hh, Hh,
        in_proj_b + l * 768, nullptr, SP * 768, nullptr, qkvb, 768, Hh, 0);
    attn_kernel<<<dim3(MT, NHh, Bb), 32, 0, stream>>>(qkvb, maskb, attb);
    gemm_kernel<<<dim3(MT, Hh / 64, Bb), 32, 0, stream>>>(
        attb, SP * Hh, Hh, wout + (size_t)l * Hh * Hh, Hh,
        out_proj_b + l * Hh, h, SP * Hh, h, nullptr, Hh, Hh, 0);
    ln_kernel<<<Bb * SP, 256, 0, stream>>>(h, ln2_s + l * Hh, ln2_b + l * Hh, hn);
    gemm_kernel<<<dim3(MT, MLPd / 64, Bb), 32, 0, stream>>>(
        hn, SP * Hh, Hh, wf1 + (size_t)l * MLPd * Hh, Hh,
        ffn_b1 + l * MLPd, nullptr, SP * MLPd, nullptr, ffb, MLPd, Hh, 1);
    gemm_kernel<<<dim3(MT, Hh / 64, Bb), 32, 0, stream>>>(
        ffb, SP * MLPd, MLPd, wf2 + (size_t)l * Hh * MLPd, MLPd,
        ffn_b2 + l * Hh, h, SP * Hh, h, nullptr, Hh, MLPd, 0);
  }
  final_ln_kernel<<<Bb, 256, 0, stream>>>(h, norm_s, norm_b, (float*)d_out);
}
